// CNNTransformerProposed_83322365542606
// MI455X (gfx1250) — matrix operand order repaired and checked by reading, not compiled, not run on hardware
//
#include <hip/hip_runtime.h>
#include <hip/hip_bf16.h>
#include <math.h>

// Problem constants (match reference)
#define BB   2
#define SEQ  2048
#define DD   256
#define HH   8
#define LL   2
#define DFFN 1024
#define DK   32
#define MM   (BB * SEQ)      // 4096 tokens
#define EPSV 1e-5f

typedef __attribute__((ext_vector_type(16))) _Float16 v16h;
typedef __attribute__((ext_vector_type(8)))  _Float16 v8h;
typedef __attribute__((ext_vector_type(8)))  float    v8f;

// ---------------------------------------------------------------------------
// Fragment loaders (WMMA 16x16x32 f16; this code's contiguous k order, same for A and B)
// A (16xK=32): lanes 0-15 hold M=lane, elems 0..15 = K 0..15;
//              lanes 16-31 hold K 16..31.  kbase = (lane>>4)*16.
// B (K=32 x 16): lane n = lane&15, kgroup = lane>>4 -> elems e = K kgroup*16+e.
// C/D (16x16 f32): lane col = lane&15, VGPR j -> row 8*(lane>>4)+j.
// ---------------------------------------------------------------------------
static __device__ __forceinline__ v16h load_a_frag(const _Float16* __restrict__ A,
                                                   int lda, int m0, int k0, int lane) {
    int m  = m0 + (lane & 15);
    int kb = k0 + ((lane >> 4) << 4);
    const _Float16* p = A + (size_t)m * lda + kb;
    return *(const v16h*)(p);
}

// Weight stored row-major [N, K]; B-frag of W^T (so C = A @ W^T)
static __device__ __forceinline__ v16h load_b_frag(const _Float16* __restrict__ Wm,
                                                   int ldk, int n0, int k0, int lane) {
    int n  = n0 + (lane & 15);
    int kb = k0 + ((lane >> 4) << 4);
    return *(const v16h*)(Wm + (size_t)n * ldk + kb);
}

// ---------------------------------------------------------------------------
// Stem: conv1d(K=3, 1->256, SAME) + "BN" + ReLU + positional encoding
// ---------------------------------------------------------------------------
__global__ __launch_bounds__(256) void stem_kernel(
    const float* __restrict__ x, const float* __restrict__ cw,
    const float* __restrict__ cb, const float* __restrict__ bg,
    const float* __restrict__ bb, const float* __restrict__ pe,
    float* __restrict__ h32, _Float16* __restrict__ h16)
{
    int idx = blockIdx.x * blockDim.x + threadIdx.x;
    if (idx >= BB * SEQ * DD) return;
    int d = idx % DD;
    int t = (idx / DD) % SEQ;
    int b = idx / (DD * SEQ);
    const float* xb = x + (size_t)b * SEQ;
    float acc = 0.0f;
#pragma unroll
    for (int kk = 0; kk < 3; ++kk) {
        int tt = t + kk - 1;
        float xv = (tt >= 0 && tt < SEQ) ? xb[tt] : 0.0f;
        acc += cw[d * 3 + kk] * xv;
    }
    acc += cb[d];
    acc = acc * (bg[d] * 0.999995000037499f) + bb[d];   // 1/sqrt(1+1e-5)
    acc = fmaxf(acc, 0.0f);
    acc += pe[(size_t)t * DD + d];
    h32[idx] = acc;
    h16[idx] = (_Float16)acc;
}

__global__ __launch_bounds__(256) void cvt_f16_kernel(const float* __restrict__ s,
                                                      _Float16* __restrict__ d, int n)
{
    int i = blockIdx.x * blockDim.x + threadIdx.x;
    if (i < n) d[i] = (_Float16)s[i];
}

// ---------------------------------------------------------------------------
// WMMA GEMM:  C[M,N] = A[M,K] @ W[N,K]^T + bias, optional relu.
// Each wave: 32x64 tile (2 A-frags x 4 B-frags -> 8 WMMAs per K=32 step,
// 0.75 fragment loads per WMMA).
// mode 0: out16/out32 row-major [M,N]
// mode 1: out16 scattered to q/k head layout [B,H,SEQ,DK]
// mode 2: out16 scattered to transposed-v layout [B,H,DK,SEQ]
// ---------------------------------------------------------------------------
__global__ __launch_bounds__(256) void gemm_wmma_kernel(
    const _Float16* __restrict__ A, const _Float16* __restrict__ Wm,
    const float* __restrict__ bias, _Float16* __restrict__ out16,
    float* __restrict__ out32, int M, int N, int K, int mode, int relu)
{
    int lane = threadIdx.x & 31;
    int wid  = (blockIdx.x * blockDim.x + threadIdx.x) >> 5;
    int mtiles = M >> 5;                       // 32-row tiles
    int nt64   = N >> 6;
    if (wid >= mtiles * nt64) return;          // wave-uniform: EXEC stays full
    int mt = wid % mtiles;
    int nt = wid / mtiles;
    int m0 = mt << 5;
    int n0 = nt << 6;

    v8f acc[2][4] = {{v8f{}, v8f{}, v8f{}, v8f{}}, {v8f{}, v8f{}, v8f{}, v8f{}}};
    for (int k0 = 0; k0 < K; k0 += 32) {
        v16h a0 = load_a_frag(A, K, m0,      k0, lane);
        v16h a1 = load_a_frag(A, K, m0 + 16, k0, lane);
        if (k0 + 32 < K) {
            __builtin_prefetch(A + (size_t)(m0 + (lane & 15)) * K + k0 + 32, 0, 1);
            __builtin_prefetch(A + (size_t)(m0 + 16 + (lane & 15)) * K + k0 + 32, 0, 1);
        }
        v16h b0 = load_b_frag(Wm, K, n0,      k0, lane);
        v16h b1 = load_b_frag(Wm, K, n0 + 16, k0, lane);
        v16h b2 = load_b_frag(Wm, K, n0 + 32, k0, lane);
        v16h b3 = load_b_frag(Wm, K, n0 + 48, k0, lane);
        acc[0][0] = __builtin_amdgcn_wmma_f32_16x16x32_f16(false, a0, false, b0, (short)0, acc[0][0], false, false);
        acc[0][1] = __builtin_amdgcn_wmma_f32_16x16x32_f16(false, a0, false, b1, (short)0, acc[0][1], false, false);
        acc[0][2] = __builtin_amdgcn_wmma_f32_16x16x32_f16(false, a0, false, b2, (short)0, acc[0][2], false, false);
        acc[0][3] = __builtin_amdgcn_wmma_f32_16x16x32_f16(false, a0, false, b3, (short)0, acc[0][3], false, false);
        acc[1][0] = __builtin_amdgcn_wmma_f32_16x16x32_f16(false, a1, false, b0, (short)0, acc[1][0], false, false);
        acc[1][1] = __builtin_amdgcn_wmma_f32_16x16x32_f16(false, a1, false, b1, (short)0, acc[1][1], false, false);
        acc[1][2] = __builtin_amdgcn_wmma_f32_16x16x32_f16(false, a1, false, b2, (short)0, acc[1][2], false, false);
        acc[1][3] = __builtin_amdgcn_wmma_f32_16x16x32_f16(false, a1, false, b3, (short)0, acc[1][3], false, false);
    }

    int half = lane >> 4;
    int col  = lane & 15;
#pragma unroll
    for (int rb = 0; rb < 2; ++rb) {
#pragma unroll
        for (int c = 0; c < 4; ++c) {
            int n = n0 + (c << 4) + col;
            float bv = bias[n];
#pragma unroll
            for (int j = 0; j < 8; ++j) {
                int m = m0 + (rb << 4) + (half << 3) + j;
                float v = acc[rb][c][j] + bv;
                if (relu) v = fmaxf(v, 0.0f);
                if (mode == 0) {
                    if (out16) out16[(size_t)m * N + n] = (_Float16)v;
                    if (out32) out32[(size_t)m * N + n] = v;
                } else {
                    int b  = m / SEQ, s = m % SEQ;
                    int hd = n / DK,  dk = n % DK;
                    if (mode == 1)
                        out16[(((size_t)b * HH + hd) * SEQ + s) * DK + dk] = (_Float16)v;
                    else
                        out16[(((size_t)b * HH + hd) * DK + dk) * SEQ + s] = (_Float16)v;
                }
            }
        }
    }
}

// ---------------------------------------------------------------------------
// Fused attention: per wave one (b, h, 16-query tile).  Streaming
// softmax(s)*sigmoid(s) with running max/denominator; QK^T and PV via WMMA.
// q,k: [B,H,SEQ,DK] f16.  vT: [B,H,DK,SEQ] f16.  ctx out: [B,SEQ,D] f16.
// ---------------------------------------------------------------------------
__global__ __launch_bounds__(256) void attn_wmma_kernel(
    const _Float16* __restrict__ q, const _Float16* __restrict__ k,
    const _Float16* __restrict__ vT, _Float16* __restrict__ ctx,
    const float* __restrict__ scale_p, const float* __restrict__ td_p, int layer)
{
    __shared__ __align__(64) _Float16 pbuf[8][16 * 32];   // per-wave P staging

    int lane   = threadIdx.x & 31;
    int wlocal = threadIdx.x >> 5;
    int wid    = blockIdx.x * 8 + wlocal;
    const int QT = SEQ / 16;
    if (wid >= BB * HH * QT) return;           // wave-uniform

    int qt = wid % QT;
    int bh = wid / QT;
    int hd = bh % HH;

    const _Float16* Qh = q  + (size_t)bh * SEQ * DK;
    const _Float16* Kh = k  + (size_t)bh * SEQ * DK;
    const _Float16* Vh = vT + (size_t)bh * DK * SEQ;

    float sc  = scale_p[layer] * 0.17677669529663687f;   // 1/sqrt(DK=32)
    float tdv = td_p[layer * HH + hd];

    v16h qf = load_a_frag(Qh, DK, qt << 4, 0, lane);

    float mrun[8], zrun[8];
#pragma unroll
    for (int j = 0; j < 8; ++j) { mrun[j] = -1e30f; zrun[j] = 0.0f; }
    v8f o0 = {}, o1 = {};

    _Float16* P = pbuf[wlocal];
    int half = lane >> 4;
    int col  = lane & 15;

    for (int kb = 0; kb < SEQ; kb += 32) {
        // K^T B-fragments: contiguous 32B per lane from K[key] row
        const _Float16* kp = Kh + (size_t)(kb + col) * DK + (half << 4);
        v16h kf0 = *(const v16h*)(kp);
        v16h kf1 = *(const v16h*)(kp + (size_t)16 * DK);

        v8f zero = {};
        v8f s0 = __builtin_amdgcn_wmma_f32_16x16x32_f16(false, qf, false, kf0, (short)0, zero, false, false);
        v8f s1 = __builtin_amdgcn_wmma_f32_16x16x32_f16(false, qf, false, kf1, (short)0, zero, false, false);

        float d0 = __expf(-tdv * (float)(kb + col));
        float d1 = __expf(-tdv * (float)(kb + 16 + col));

        float sv0[8], sv1[8];
#pragma unroll
        for (int j = 0; j < 8; ++j) { sv0[j] = s0[j] * sc * d0; sv1[j] = s1[j] * sc * d1; }

#pragma unroll
        for (int j = 0; j < 8; ++j) {
            // block row-max over 32 keys (reduce across 16-lane half)
            float bm = fmaxf(sv0[j], sv1[j]);
#pragma unroll
            for (int off = 1; off < 16; off <<= 1)
                bm = fmaxf(bm, __shfl_xor(bm, off, 32));

            float mn    = fmaxf(mrun[j], bm);
            float alpha = __expf(mrun[j] - mn);
            mrun[j] = mn;

            float e0 = __expf(sv0[j] - mn);
            float e1 = __expf(sv1[j] - mn);
            float g0 = e0 / (1.0f + __expf(-sv0[j]));   // exp * sigmoid
            float g1 = e1 / (1.0f + __expf(-sv1[j]));

            float zb = e0 + e1;                          // softmax denom uses exp only
#pragma unroll
            for (int off = 1; off < 16; off <<= 1)
                zb += __shfl_xor(zb, off, 32);
            zrun[j] = zrun[j] * alpha + zb;

            o0[j] *= alpha;
            o1[j] *= alpha;

            int row = (half << 3) + j;                   // C-layout row
            P[row * 32 + col]      = (_Float16)g0;
            P[row * 32 + col + 16] = (_Float16)g1;
        }

        asm volatile("s_wait_dscnt 0" ::: "memory");     // LDS store->load fence

        // Re-read P in A-fragment layout, then two PV WMMAs
        {
            const _Float16* pp = P + col * 32 + (half << 4);
            v16h pf = *(const v16h*)(pp);

            const _Float16* vp0 = Vh + (size_t)col * SEQ + kb + (half << 4);
            const _Float16* vp1 = Vh + (size_t)(col + 16) * SEQ + kb + (half << 4);
            v16h vf0 = *(const v16h*)vp0;
            v16h vf1 = *(const v16h*)vp1;

            o0 = __builtin_amdgcn_wmma_f32_16x16x32_f16(false, pf, false, vf0, (short)0, o0, false, false);
            o1 = __builtin_amdgcn_wmma_f32_16x16x32_f16(false, pf, false, vf1, (short)0, o1, false, false);
        }
    }

    int b = bh / HH;
#pragma unroll
    for (int j = 0; j < 8; ++j) {
        int s = (qt << 4) + (half << 3) + j;
        float inz = 1.0f / zrun[j];
        size_t base = ((size_t)b * SEQ + s) * DD + hd * DK;
        ctx[base + col]      = (_Float16)(o0[j] * inz);
        ctx[base + col + 16] = (_Float16)(o1[j] * inz);
    }
}

// ---------------------------------------------------------------------------
// h = LayerNorm(h + r) * g + b   (one block per token, D=256 threads)
// ---------------------------------------------------------------------------
__global__ __launch_bounds__(256) void add_ln_kernel(
    float* __restrict__ h32, _Float16* __restrict__ h16,
    const float* __restrict__ r, const float* __restrict__ g,
    const float* __restrict__ bta)
{
    __shared__ float red[256];
    int t = blockIdx.x;
    int d = threadIdx.x;
    size_t base = (size_t)t * DD;
    float z = h32[base + d] + r[base + d];

    red[d] = z;
    __syncthreads();
    for (int off = 128; off > 0; off >>= 1) {
        if (d < off) red[d] += red[d + off];
        __syncthreads();
    }
    float mean = red[0] * (1.0f / DD);
    __syncthreads();

    float dz = z - mean;
    red[d] = dz * dz;
    __syncthreads();
    for (int off = 128; off > 0; off >>= 1) {
        if (d < off) red[d] += red[d + off];
        __syncthreads();
    }
    float var = red[0] * (1.0f / DD);

    float y = dz * rsqrtf(var + EPSV) * g[d] + bta[d];
    h32[base + d] = y;
    h16[base + d] = (_Float16)y;
}

// ---------------------------------------------------------------------------
// out[b] = 0.5 * sum_d h[b, SEQ-1, d] * outW[d] + outb[0]   (drop-connect 0.5)
// ---------------------------------------------------------------------------
__global__ __launch_bounds__(256) void head_kernel(
    const float* __restrict__ h32, const float* __restrict__ outW,
    const float* __restrict__ outb, float* __restrict__ out)
{
    __shared__ float red[256];
    int b = blockIdx.x;
    int d = threadIdx.x;
    const float* last = h32 + ((size_t)b * SEQ + (SEQ - 1)) * DD;
    red[d] = last[d] * outW[d];
    __syncthreads();
    for (int off = 128; off > 0; off >>= 1) {
        if (d < off) red[d] += red[d + off];
        __syncthreads();
    }
    if (d == 0) out[b] = red[0] * 0.5f + outb[0];
}

// ---------------------------------------------------------------------------
extern "C" void kernel_launch(void* const* d_in, const int* in_sizes, int n_in,
                              void* d_out, int out_size, void* d_ws, size_t ws_size,
                              hipStream_t stream)
{
    const float* x      = (const float*)d_in[0];
    const float* conv_w = (const float*)d_in[1];
    const float* conv_b = (const float*)d_in[2];
    const float* bn_g   = (const float*)d_in[3];
    const float* bn_b   = (const float*)d_in[4];
    const float* pe     = (const float*)d_in[5];
    const float* qW     = (const float*)d_in[6];
    const float* qb     = (const float*)d_in[7];
    const float* kW     = (const float*)d_in[8];
    const float* kbv    = (const float*)d_in[9];
    const float* vW     = (const float*)d_in[10];
    const float* vb     = (const float*)d_in[11];
    const float* oW     = (const float*)d_in[12];
    const float* ob     = (const float*)d_in[13];
    const float* scale  = (const float*)d_in[14];
    const float* td     = (const float*)d_in[15];
    const float* ln1g   = (const float*)d_in[16];
    const float* ln1b   = (const float*)d_in[17];
    const float* f1W    = (const float*)d_in[18];
    const float* f1b    = (const float*)d_in[19];
    const float* f2W    = (const float*)d_in[20];
    const float* f2b    = (const float*)d_in[21];
    const float* ln2g   = (const float*)d_in[22];
    const float* ln2b   = (const float*)d_in[23];
    const float* outW   = (const float*)d_in[24];
    const float* outb   = (const float*)d_in[25];

    // Workspace carve-out (256B aligned)
    char* p = (char*)d_ws;
    auto carve = [&](size_t bytes) -> char* {
        char* r = p;
        p += (bytes + 255) & ~(size_t)255;
        return r;
    };
    float*    h32   = (float*)   carve((size_t)MM * DD * 4);
    _Float16* h16   = (_Float16*)carve((size_t)MM * DD * 2);
    _Float16* qW16  = (_Float16*)carve((size_t)LL * DD * DD * 2);
    _Float16* kW16  = (_Float16*)carve((size_t)LL * DD * DD * 2);
    _Float16* vW16  = (_Float16*)carve((size_t)LL * DD * DD * 2);
    _Float16* oW16  = (_Float16*)carve((size_t)LL * DD * DD * 2);
    _Float16* f1W16 = (_Float16*)carve((size_t)LL * DFFN * DD * 2);
    _Float16* f2W16 = (_Float16*)carve((size_t)LL * DD * DFFN * 2);
    _Float16* q16   = (_Float16*)carve((size_t)MM * DD * 2);
    _Float16* k16   = (_Float16*)carve((size_t)MM * DD * 2);
    _Float16* vT16  = (_Float16*)carve((size_t)MM * DD * 2);
    _Float16* ctx16 = (_Float16*)carve((size_t)MM * DD * 2);
    _Float16* a16   = (_Float16*)carve((size_t)MM * DFFN * 2);
    float*    r32   = (float*)   carve((size_t)MM * DD * 4);
    (void)ws_size; (void)in_sizes; (void)n_in; (void)out_size;

    // Weight conversion f32 -> f16
    {
        int n1 = LL * DD * DD;
        int n2 = LL * DFFN * DD;
        cvt_f16_kernel<<<(n1 + 255) / 256, 256, 0, stream>>>(qW, qW16, n1);
        cvt_f16_kernel<<<(n1 + 255) / 256, 256, 0, stream>>>(kW, kW16, n1);
        cvt_f16_kernel<<<(n1 + 255) / 256, 256, 0, stream>>>(vW, vW16, n1);
        cvt_f16_kernel<<<(n1 + 255) / 256, 256, 0, stream>>>(oW, oW16, n1);
        cvt_f16_kernel<<<(n2 + 255) / 256, 256, 0, stream>>>(f1W, f1W16, n2);
        cvt_f16_kernel<<<(n2 + 255) / 256, 256, 0, stream>>>(f2W, f2W16, n2);
    }

    // Stem
    stem_kernel<<<(MM * DD + 255) / 256, 256, 0, stream>>>(
        x, conv_w, conv_b, bn_g, bn_b, pe, h32, h16);

    auto gemm_grid = [](int M, int N) {
        int waves = (M >> 5) * (N >> 6);
        return (waves + 7) / 8;
    };

    for (int l = 0; l < LL; ++l) {
        const _Float16* qw  = qW16  + (size_t)l * DD * DD;
        const _Float16* kw  = kW16  + (size_t)l * DD * DD;
        const _Float16* vw  = vW16  + (size_t)l * DD * DD;
        const _Float16* ow  = oW16  + (size_t)l * DD * DD;
        const _Float16* f1w = f1W16 + (size_t)l * DFFN * DD;
        const _Float16* f2w = f2W16 + (size_t)l * DD * DFFN;

        // Q/K/V projections (scattered into head layouts)
        gemm_wmma_kernel<<<gemm_grid(MM, DD), 256, 0, stream>>>(
            h16, qw, qb + l * DD, q16, nullptr, MM, DD, DD, 1, 0);
        gemm_wmma_kernel<<<gemm_grid(MM, DD), 256, 0, stream>>>(
            h16, kw, kbv + l * DD, k16, nullptr, MM, DD, DD, 1, 0);
        gemm_wmma_kernel<<<gemm_grid(MM, DD), 256, 0, stream>>>(
            h16, vw, vb + l * DD, vT16, nullptr, MM, DD, DD, 2, 0);

        // Fused attention
        attn_wmma_kernel<<<(BB * HH * (SEQ / 16)) / 8, 256, 0, stream>>>(
            q16, k16, vT16, ctx16, scale, td, l);

        // Output projection (f32 result) + residual LayerNorm
        gemm_wmma_kernel<<<gemm_grid(MM, DD), 256, 0, stream>>>(
            ctx16, ow, ob + l * DD, nullptr, r32, MM, DD, DD, 0, 0);
        add_ln_kernel<<<MM, 256, 0, stream>>>(h32, h16, r32, ln1g + l * DD, ln1b + l * DD);

        // FFN
        gemm_wmma_kernel<<<gemm_grid(MM, DFFN), 256, 0, stream>>>(
            h16, f1w, f1b + l * DFFN, a16, nullptr, MM, DFFN, DD, 0, 1);
        gemm_wmma_kernel<<<gemm_grid(MM, DD), 256, 0, stream>>>(
            a16, f2w, f2b + l * DD, nullptr, r32, MM, DD, DFFN, 0, 0);
        add_ln_kernel<<<MM, 256, 0, stream>>>(h32, h16, r32, ln2g + l * DD, ln2b + l * DD);
    }

    head_kernel<<<BB, 256, 0, stream>>>(h32, outW, outb, (float*)d_out);
}
